// TourismGNN_25632364822987
// MI455X (gfx1250) — compile-verified, matrix operand order checked by reading, not run on hardware
//
#include <hip/hip_runtime.h>
#include <hip/hip_bf16.h>

typedef __attribute__((ext_vector_type(16))) _Float16 v16h;
typedef __attribute__((ext_vector_type(8)))  float    v8f;

#define NFEAT   128
#define HID     16

// ---------------------------------------------------------------------------
// K0: deg[i] = 1.0f  (self-loop contribution to degree)
// ---------------------------------------------------------------------------
__global__ void k_deg_init(float* __restrict__ deg, int N) {
    int i = blockIdx.x * blockDim.x + threadIdx.x;
    if (i < N) deg[i] = 1.0f;
}

// ---------------------------------------------------------------------------
// K1: deg[col[e]] += 1  for every edge
// ---------------------------------------------------------------------------
__global__ void k_deg_count(const long long* __restrict__ ec, float* __restrict__ deg, int E) {
    int e = blockIdx.x * blockDim.x + threadIdx.x;
    if (e < E) atomicAdd(&deg[(int)ec[e]], 1.0f);
}

// ---------------------------------------------------------------------------
// K2: dinv[i] = rsqrt(deg[i])   (deg >= 1 always due to self loops)
// ---------------------------------------------------------------------------
__global__ void k_rsqrt(float* __restrict__ deg, int N) {
    int i = blockIdx.x * blockDim.x + threadIdx.x;
    if (i < N) deg[i] = rsqrtf(deg[i]);
}

// ---------------------------------------------------------------------------
// K3: h = x @ W1 via V_WMMA_F32_16X16X32_F16 (f32 accumulate).
//     One wave computes a 16x16 output tile (16 rows x all 16 hidden).
//     Epilogue also seeds out1 with the self-loop term dinv^2 * h.
//
//  A 16x32 f16 layout (per ISA 7.12.2): lane l -> M = l&15, half = l>>4;
//    v16h elems 0..7  = K in [8*half, 8*half+7]
//    v16h elems 8..15 = K in [16+8*half, 16+8*half+7]
//  B 32x16 mirrors A with N = l&15.
//  C/D f32: lane l holds column n=l&15 of rows M = 8*(l>>4)+v, v=0..7.
// ---------------------------------------------------------------------------
__global__ void k_gemm_wmma(const float* __restrict__ x,
                            const float* __restrict__ W1,
                            const float* __restrict__ dinv,
                            float* __restrict__ h,
                            float* __restrict__ out1,
                            int nTiles) {
    const int lane = threadIdx.x & 31;
    const int wave = threadIdx.x >> 5;
    const int tile = blockIdx.x * (blockDim.x >> 5) + wave;
    if (tile >= nTiles) return;           // wave-uniform: EXEC all-1s inside active waves

    const int m0   = tile * 16;
    const int nn   = lane & 15;           // N column (B/C) or M row (A)
    const int half = lane >> 4;
    const int mrow = m0 + nn;             // A-matrix row for this lane

    v8f c = {};
#pragma unroll
    for (int s = 0; s < 4; ++s) {
        const int k0 = s * 32;
        v16h a, b;
        const float* xp = x + (size_t)mrow * NFEAT + k0 + 8 * half;
#pragma unroll
        for (int j = 0; j < 8; ++j) {
            a[j]     = (_Float16)xp[j];
            a[j + 8] = (_Float16)xp[16 + j];
        }
#pragma unroll
        for (int j = 0; j < 8; ++j) {
            b[j]     = (_Float16)W1[(k0 + 8 * half + j)      * HID + nn];
            b[j + 8] = (_Float16)W1[(k0 + 16 + 8 * half + j) * HID + nn];
        }
        c = __builtin_amdgcn_wmma_f32_16x16x32_f16(
                /*neg_a=*/false, a, /*neg_b=*/false, b,
                /*c_mod=*/(short)0, c, /*reuse_a=*/false, /*reuse_b=*/false);
    }

#pragma unroll
    for (int v = 0; v < 8; ++v) {
        const int m = m0 + 8 * half + v;
        const float d  = dinv[m];
        const float hv = c[v];
        h[(size_t)m * HID + nn]    = hv;
        out1[(size_t)m * HID + nn] = d * d * hv;   // self-loop seed
    }
}

// ---------------------------------------------------------------------------
// K4: layer-1 edge scatter: out1[col] += norm * h[row]  (16 features/edge)
// ---------------------------------------------------------------------------
__global__ void k_edge_scatter1(const long long* __restrict__ er,
                                const long long* __restrict__ ec,
                                const float* __restrict__ dinv,
                                const float* __restrict__ h,
                                float* __restrict__ out1, int E) {
    int e = blockIdx.x * blockDim.x + threadIdx.x;
    if (e >= E) return;
    const int r = (int)er[e];
    const int c = (int)ec[e];
    const float nrm = dinv[r] * dinv[c];
    const float4* hp = (const float4*)(h + (size_t)r * HID);
    float* op = out1 + (size_t)c * HID;
#pragma unroll
    for (int q = 0; q < 4; ++q) {
        const float4 hv = hp[q];
        atomicAdd(op + 4 * q + 0, nrm * hv.x);
        atomicAdd(op + 4 * q + 1, nrm * hv.y);
        atomicAdd(op + 4 * q + 2, nrm * hv.z);
        atomicAdd(op + 4 * q + 3, nrm * hv.w);
    }
}

// ---------------------------------------------------------------------------
// K5: g = relu(out1 + b1) @ W2 ; out seeded with b2 + dinv^2 * g (self loop)
// ---------------------------------------------------------------------------
__global__ void k_finalize1(const float* __restrict__ out1,
                            const float* __restrict__ b1,
                            const float* __restrict__ W2,
                            const float* __restrict__ b2,
                            const float* __restrict__ dinv,
                            float* __restrict__ g,
                            float* __restrict__ out, int N) {
    int i = blockIdx.x * blockDim.x + threadIdx.x;
    if (i >= N) return;
    const float* op = out1 + (size_t)i * HID;
    float acc = 0.0f;
#pragma unroll
    for (int f = 0; f < HID; ++f) {
        float v = op[f] + b1[f];
        v = fmaxf(v, 0.0f);
        acc = fmaf(v, W2[f], acc);
    }
    g[i] = acc;
    const float d = dinv[i];
    out[i] = b2[0] + d * d * acc;
}

// ---------------------------------------------------------------------------
// K6: layer-2 edge scatter: out[col] += norm * g[row]
// ---------------------------------------------------------------------------
__global__ void k_edge_scatter2(const long long* __restrict__ er,
                                const long long* __restrict__ ec,
                                const float* __restrict__ dinv,
                                const float* __restrict__ g,
                                float* __restrict__ out, int E) {
    int e = blockIdx.x * blockDim.x + threadIdx.x;
    if (e >= E) return;
    const int r = (int)er[e];
    const int c = (int)ec[e];
    atomicAdd(&out[c], dinv[r] * dinv[c] * g[r]);
}

// ---------------------------------------------------------------------------
extern "C" void kernel_launch(void* const* d_in, const int* in_sizes, int n_in,
                              void* d_out, int out_size, void* d_ws, size_t ws_size,
                              hipStream_t stream) {
    const float*     x     = (const float*)d_in[0];
    const long long* edges = (const long long*)d_in[1];
    const float*     W1    = (const float*)d_in[2];
    const float*     b1    = (const float*)d_in[3];
    const float*     W2    = (const float*)d_in[4];
    const float*     b2    = (const float*)d_in[5];
    float*           out   = (float*)d_out;

    const int N = in_sizes[0] / NFEAT;   // 100000
    const int E = in_sizes[1] / 2;       // 3200000
    const long long* er = edges;         // source (row)
    const long long* ec = edges + E;     // target (col)

    // workspace layout (floats): dinv | h[N*16] | out1[N*16] | g[N]
    float* ws   = (float*)d_ws;
    float* deg  = ws;                               // reused as dinv after K2
    float* h    = ws + N;
    float* out1 = h + (size_t)N * HID;
    float* g    = out1 + (size_t)N * HID;

    const int TB = 256;
    const int nBlkN = (N + TB - 1) / TB;
    const int nBlkE = (E + TB - 1) / TB;

    // 1. degree (with self loops)
    k_deg_init <<<nBlkN, TB, 0, stream>>>(deg, N);
    k_deg_count<<<nBlkE, TB, 0, stream>>>(ec, deg, E);
    k_rsqrt    <<<nBlkN, TB, 0, stream>>>(deg, N);

    // 2. h = x @ W1 (WMMA), seed out1 with self-loop term
    const int nTiles = (N + 15) / 16;                     // 6250
    const int wavesPerBlock = TB / 32;                    // 8
    const int nBlkG = (nTiles + wavesPerBlock - 1) / wavesPerBlock;
    k_gemm_wmma<<<nBlkG, TB, 0, stream>>>(x, W1, deg, h, out1, nTiles);

    // 3. layer-1 aggregation
    k_edge_scatter1<<<nBlkE, TB, 0, stream>>>(er, ec, deg, h, out1, E);

    // 4. relu + layer-2 linear, seed out with bias + self loop
    k_finalize1<<<nBlkN, TB, 0, stream>>>(out1, b1, W2, b2, deg, g, out, N);

    // 5. layer-2 aggregation
    k_edge_scatter2<<<nBlkE, TB, 0, stream>>>(er, ec, deg, g, out, E);
}
